// GraphConvBlock_42039139893787
// MI455X (gfx1250) — compile-verified, matrix operand order checked by reading, not run on hardware
//
#include <hip/hip_runtime.h>
#include <hip/hip_bf16.h>
#include <math.h>

#define NN   50000
#define DD   128
#define HH   4
#define HD   512
#define EE   800000
#define ETOT 850000
#define MT   (NN / 16)   // 3125 row tiles

typedef __bf16 bf16_t;
typedef __attribute__((ext_vector_type(16))) __bf16 v16bf;
typedef __attribute__((ext_vector_type(8)))  float  v8f;

// ---------------------------------------------------------------- utilities
__global__ __launch_bounds__(256) void fill_kernel(float* __restrict__ p, float v, int n) {
    int i = blockIdx.x * 256 + threadIdx.x;
    if (i < n) p[i] = v;
}

__device__ __forceinline__ void atomicMaxF(float* addr, float val) {
    unsigned int* ua = (unsigned int*)addr;
    unsigned int old = *ua;
    while (__uint_as_float(old) < val) {
        unsigned int assumed = old;
        old = atomicCAS(ua, assumed, __float_as_uint(val));
        if (old == assumed) break;
    }
}

// ------------------------------------------------- W -> bf16 hi/lo fragments
// Fragment-native layout: idx = ((ntile*4 + kchunk)*32 + lane)*16 + e
// lane: n = ntile*16 + (lane&15), k-half = (lane>>4)*8
// e:    k = kchunk*32 + (e&7) + (e>=8 ? 16 : 0) + k-half
__global__ __launch_bounds__(256)
void wconv_kernel(const float* __restrict__ W, bf16_t* __restrict__ hi, bf16_t* __restrict__ lo) {
    int tid = blockIdx.x * 256 + threadIdx.x;     // 0..65535
    int e = tid & 15;
    int l = (tid >> 4) & 31;
    int c = (tid >> 9) & 3;
    int t = tid >> 11;                            // ntile 0..31
    int n = t * 16 + (l & 15);
    int k = c * 32 + (e & 7) + ((e >> 3) << 4) + ((l >> 4) << 3);
    float w = W[k * HD + n];
    bf16_t h = (bf16_t)w;
    hi[tid] = h;
    lo[tid] = (bf16_t)(w - (float)h);
}

// ------------------------------------------------- GEMM  h = x @ W  (WMMA)
// block = 256 (8 waves). Block owns one 16-row tile of x; wave w owns output
// columns [w*64, w*64+64). fp32 emulated via 3x bf16 wmma (hi*hi+hi*lo+lo*hi).
__global__ __launch_bounds__(256)
void gemm_kernel(const float* __restrict__ X, const bf16_t* __restrict__ Whi,
                 const bf16_t* __restrict__ Wlo, float* __restrict__ Hout) {
    __shared__ float xs[16 * 128];   // 8 KB x tile
    int mbase = blockIdx.x * 16;
    for (int i = threadIdx.x; i < 16 * 128; i += 256)
        xs[i] = X[(size_t)mbase * 128 + i];
    __syncthreads();

    int lane = threadIdx.x & 31;
    int wave = threadIdx.x >> 5;
    int m = lane & 15;
    int khalf = (lane >> 4) << 3;

    // A fragments for K=128 (4 chunks of 32), hi/lo split
    v16bf ahi[4], alo[4];
#pragma unroll
    for (int c = 0; c < 4; ++c) {
#pragma unroll
        for (int e = 0; e < 16; ++e) {
            int k = c * 32 + (e & 7) + ((e >> 3) << 4) + khalf;
            float v = xs[m * 128 + k];
            bf16_t h = (bf16_t)v;
            ahi[c][e] = h;
            alo[c][e] = (bf16_t)(v - (float)h);
        }
    }

#pragma unroll
    for (int nt = 0; nt < 4; ++nt) {
        int ntg = wave * 4 + nt;                  // global 16-col tile 0..31
        v8f acc = {};
#pragma unroll
        for (int c = 0; c < 4; ++c) {
            int base = ((ntg * 4 + c) * 32 + lane) * 16;
            v16bf bhi = *(const v16bf*)(Whi + base);
            v16bf blo = *(const v16bf*)(Wlo + base);
            acc = __builtin_amdgcn_wmma_f32_16x16x32_bf16(false, ahi[c], false, bhi,
                                                          (short)0, acc, false, false);
            acc = __builtin_amdgcn_wmma_f32_16x16x32_bf16(false, alo[c], false, bhi,
                                                          (short)0, acc, false, false);
            acc = __builtin_amdgcn_wmma_f32_16x16x32_bf16(false, ahi[c], false, blo,
                                                          (short)0, acc, false, false);
        }
        // C layout: lane n = lane&15; VGPR r -> M = r + (lane>=16 ? 8 : 0)
        int n = ntg * 16 + (lane & 15);
        int mrow = mbase + ((lane >> 4) << 3);
#pragma unroll
        for (int r = 0; r < 8; ++r)
            Hout[(size_t)(mrow + r) * HD + n] = acc[r];
    }
}

// ------------------------------------------------- per-node attention scalars
__global__ __launch_bounds__(128)
void att_kernel(const float* __restrict__ Hm, const float* __restrict__ as_,
                const float* __restrict__ ad_, float* __restrict__ asrc,
                float* __restrict__ adst) {
    __shared__ float red[8][128];
    int n = blockIdx.x, t = threadIdx.x;
#pragma unroll
    for (int hh = 0; hh < HH; ++hh) {
        float hv = Hm[(size_t)n * HD + hh * DD + t];
        red[hh][t]      = hv * as_[hh * DD + t];
        red[hh + 4][t]  = hv * ad_[hh * DD + t];
    }
    __syncthreads();
    for (int s = 64; s > 0; s >>= 1) {
        if (t < s)
#pragma unroll
            for (int r = 0; r < 8; ++r) red[r][t] += red[r][t + s];
        __syncthreads();
    }
    if (t < HH) { asrc[n * HH + t] = red[t][0]; adst[n * HH + t] = red[t + 4][0]; }
}

// ------------------------------------------------- edge logits + segment max
__global__ __launch_bounds__(256)
void logits_kernel(const int* __restrict__ ei, const float* __restrict__ asrc,
                   const float* __restrict__ adst, float* __restrict__ lg,
                   float* __restrict__ lmax) {
    int e = blockIdx.x * 256 + threadIdx.x;
    if (e >= ETOT) return;
    int s = (e < EE) ? ei[e] : (e - EE);
    int d = (e < EE) ? ei[EE + e] : (e - EE);
#pragma unroll
    for (int hh = 0; hh < HH; ++hh) {
        float v = asrc[s * HH + hh] + adst[d * HH + hh];
        v = v > 0.f ? v : 0.2f * v;               // leaky_relu(0.2)
        lg[(size_t)e * HH + hh] = v;
        atomicMaxF(&lmax[d * HH + hh], v);
    }
}

// ------------------------------------------------- p = exp(l - max); denom
__global__ __launch_bounds__(256)
void pden_kernel(const int* __restrict__ ei, float* __restrict__ lg,
                 const float* __restrict__ lmax, float* __restrict__ den) {
    int e = blockIdx.x * 256 + threadIdx.x;
    if (e >= ETOT) return;
    int d = (e < EE) ? ei[EE + e] : (e - EE);
#pragma unroll
    for (int hh = 0; hh < HH; ++hh) {
        float p = __expf(lg[(size_t)e * HH + hh] - lmax[d * HH + hh]);
        lg[(size_t)e * HH + hh] = p;
        atomicAdd(&den[d * HH + hh], p);
    }
}

// ------------------------------------------------- weighted scatter-add agg
// 2 edges per block; 128 threads per edge (thread = channel, loops heads)
__global__ __launch_bounds__(256)
void agg_kernel(const int* __restrict__ ei, const float* __restrict__ Hm,
                const float* __restrict__ pbuf, const float* __restrict__ den,
                float* __restrict__ agg) {
    int e = blockIdx.x * 2 + (threadIdx.x >> 7);
    if (e >= ETOT) return;
    int c = threadIdx.x & 127;
    int s = (e < EE) ? ei[e] : (e - EE);
    int d = (e < EE) ? ei[EE + e] : (e - EE);
#pragma unroll
    for (int hh = 0; hh < HH; ++hh) {
        float alpha = pbuf[(size_t)e * HH + hh] / den[d * HH + hh];
        float v = alpha * Hm[(size_t)s * HD + hh * DD + c];
        atomicAdd(&agg[(size_t)d * HD + hh * DD + c], v);
    }
}

// ------------------------------------------------- head-mean + bias + column stats
__global__ __launch_bounds__(128)
void meanstats_kernel(const float* __restrict__ agg, const float* __restrict__ bias,
                      float* __restrict__ y, float* __restrict__ cst) {
    int c = threadIdx.x;
    float s = 0.f, s2 = 0.f;
    float b = bias[c];
    for (int n = blockIdx.x; n < NN; n += gridDim.x) {
        const float* a = agg + (size_t)n * HD;
        float v = 0.25f * (a[c] + a[DD + c] + a[2 * DD + c] + a[3 * DD + c]) + b;
        y[(size_t)n * DD + c] = v;
        s += v; s2 += v * v;
    }
    atomicAdd(&cst[c], s);
    atomicAdd(&cst[DD + c], s2);
}

// ------------------------------------------------- batchnorm + relu (in place)
__global__ __launch_bounds__(256)
void bn_kernel(float* __restrict__ y, const float* __restrict__ cst,
               const float* __restrict__ g, const float* __restrict__ be) {
    int i = blockIdx.x * 256 + threadIdx.x;
    if (i >= NN * DD) return;
    int c = i & (DD - 1);
    float mu  = cst[c] * (1.0f / NN);
    float var = cst[DD + c] * (1.0f / NN) - mu * mu;
    float v = (y[i] - mu) * rsqrtf(var + 1e-5f) * g[c] + be[c];
    y[i] = v > 0.f ? v : 0.f;
}

// ---------------------------------------------------------------- host side
struct Scratch {
    float *h, *agg, *lg, *lmax, *den, *asrc, *adst, *x2, *cst;
    bf16_t *whi, *wlo;
};

static void run_layer(const float* xin, const int* ei, const float* W,
                      const float* as_, const float* ad_, const float* b,
                      const float* g, const float* be, float* yout,
                      const Scratch& S, hipStream_t stream) {
    wconv_kernel<<<256, 256, 0, stream>>>(W, S.whi, S.wlo);
    gemm_kernel<<<MT, 256, 0, stream>>>(xin, S.whi, S.wlo, S.h);
    att_kernel<<<NN, 128, 0, stream>>>(S.h, as_, ad_, S.asrc, S.adst);

    fill_kernel<<<(NN * HD + 255) / 256, 256, 0, stream>>>(S.agg, 0.f, NN * HD);
    fill_kernel<<<(NN * HH + 255) / 256, 256, 0, stream>>>(S.lmax, -1e38f, NN * HH);
    fill_kernel<<<(NN * HH + 255) / 256, 256, 0, stream>>>(S.den, 0.f, NN * HH);
    fill_kernel<<<1, 256, 0, stream>>>(S.cst, 0.f, 2 * DD);

    logits_kernel<<<(ETOT + 255) / 256, 256, 0, stream>>>(ei, S.asrc, S.adst, S.lg, S.lmax);
    pden_kernel<<<(ETOT + 255) / 256, 256, 0, stream>>>(ei, S.lg, S.lmax, S.den);
    agg_kernel<<<(ETOT + 1) / 2, 256, 0, stream>>>(ei, S.h, S.lg, S.den, S.agg);

    meanstats_kernel<<<512, 128, 0, stream>>>(S.agg, b, yout, S.cst);
    bn_kernel<<<(NN * DD + 255) / 256, 256, 0, stream>>>(yout, S.cst, g, be);
}

extern "C" void kernel_launch(void* const* d_in, const int* in_sizes, int n_in,
                              void* d_out, int out_size, void* d_ws, size_t ws_size,
                              hipStream_t stream) {
    const float* x  = (const float*)d_in[0];
    const int*   ei = (const int*)d_in[1];
    const float* W1 = (const float*)d_in[2];
    const float* as1 = (const float*)d_in[3];
    const float* ad1 = (const float*)d_in[4];
    const float* b1 = (const float*)d_in[5];
    const float* g1 = (const float*)d_in[6];
    const float* be1 = (const float*)d_in[7];
    const float* W2 = (const float*)d_in[8];
    const float* as2 = (const float*)d_in[9];
    const float* ad2 = (const float*)d_in[10];
    const float* b2 = (const float*)d_in[11];
    const float* g2 = (const float*)d_in[12];
    const float* be2 = (const float*)d_in[13];

    float* ws = (float*)d_ws;
    Scratch S;
    size_t off = 0;
    S.h    = ws + off; off += (size_t)NN * HD;       // 25.6 M
    S.agg  = ws + off; off += (size_t)NN * HD;       // 25.6 M
    S.lg   = ws + off; off += (size_t)ETOT * HH;     // 3.4 M
    S.lmax = ws + off; off += (size_t)NN * HH;
    S.den  = ws + off; off += (size_t)NN * HH;
    S.asrc = ws + off; off += (size_t)NN * HH;
    S.adst = ws + off; off += (size_t)NN * HH;
    S.x2   = ws + off; off += (size_t)NN * DD;       // 6.4 M
    S.cst  = ws + off; off += 256;
    S.whi  = (bf16_t*)(ws + off);
    S.wlo  = S.whi + 65536;                          // 256 KB total bf16

    // Layer 1: x -> x2
    run_layer(x, ei, W1, as1, ad1, b1, g1, be1, S.x2, S, stream);
    // Layer 2: x2 -> d_out
    run_layer(S.x2, ei, W2, as2, ad2, b2, g2, be2, (float*)d_out, S, stream);
}
